// MultiHead_SelfAttention_5600637354003
// MI455X (gfx1250) — matrix operand order repaired and checked by reading, not compiled, not run on hardware
//
#include <hip/hip_runtime.h>

typedef __attribute__((ext_vector_type(16))) _Float16 v16h;
typedef __attribute__((ext_vector_type(8)))  float    v8f;
typedef __attribute__((ext_vector_type(4)))  float    v4f;
typedef __attribute__((ext_vector_type(4)))  unsigned v4u;
typedef __attribute__((ext_vector_type(2)))  unsigned v2u;
typedef __attribute__((ext_vector_type(4)))  int      v4i;

#define LEN    2048
#define DD     512
#define NBB    4            // batches per block
#define MROWS  128          // NBB * 32
#define XPITCH 520          // halfs per LDS row of x (512 + 8 pad)

// ---- CDNA5 async global->LDS copy (guarded; falls back to ld+ds_store) ----
#if defined(__has_builtin)
#if __has_builtin(__builtin_amdgcn_global_load_async_to_lds_b128)
#define HAVE_ASYNC_LDS 1
#endif
#endif
#ifndef HAVE_ASYNC_LDS
#define HAVE_ASYNC_LDS 0
#endif

#if HAVE_ASYNC_LDS
typedef __attribute__((address_space(1))) v4i* gas_ptr;   // global int4*
typedef __attribute__((address_space(3))) v4i* las_ptr;   // LDS int4*
// generic->AS1: same 64-bit address; generic->AS3: low 32 bits are the LDS offset (ISA 10.2)
#define TO_GAS(p) ((gas_ptr)(unsigned long long)(const char*)(p))
#define TO_LAS(p) ((las_ptr)(unsigned int)(unsigned long long)(const char*)(p))
#endif

__device__ __forceinline__ void wait_async_copies() {
#if HAVE_ASYNC_LDS
#if __has_builtin(__builtin_amdgcn_s_wait_asynccnt)
    __builtin_amdgcn_s_wait_asynccnt(0);
#else
    asm volatile("s_wait_asynccnt 0x0" ::: "memory");
#endif
#endif
}

// ---- LDS layout (bytes) ----
#define OFF_X    0
#define SZ_X     (MROWS * XPITCH * 2)        // 133120
#define OFF_B    (OFF_X + SZ_X)
#define SZ_B     (2 * 16 * 512 * 2)          // 32768 : double-buffered 16 frag slots
#define OFF_F    (OFF_B + SZ_B)              // 165888
#define SZ_F     (256 * 8 * 4)               // 8192  : factor[a]
#define OFF_KF   (OFF_F + SZ_F)
#define SZ_KF    (NBB * 64 * 8 * 4)          // 8192
#define OFF_VF   (OFF_KF + SZ_KF)
#define OFF_SC   (OFF_VF + SZ_KF)
#define SZ_SC    (NBB * 256 * 8 * 4)         // 32768
#define OFF_BIAS (OFF_SC + SZ_SC)
#define SZ_BIAS  (3 * 512 * 4)               // 6144
#define SMEM_TOTAL (OFF_BIAS + SZ_BIAS)      // 229376 B (<= 320KB WGP LDS)

// ---------------------------------------------------------------------------
// Prep: repack Wk,Wv,Wq (f32 [e][d]) into f16 WMMA-B fragments.
// Fragment (p, nt, kt): lane holds halfs j=0..15 of B[k][n] = W[n][k] with
//   n = nt*16 + (lane&15),  k = kt*32 + j + 16*(lane>>4)
// stored contiguously: ((p*32+nt)*16+kt)*32*16 + lane*16 + j  (halfs)
// ---------------------------------------------------------------------------
__global__ __launch_bounds__(256) void prep_weights(
    const float* __restrict__ Wk, const float* __restrict__ Wv,
    const float* __restrict__ Wq, _Float16* __restrict__ Bpre)
{
    int id   = blockIdx.x * 256 + threadIdx.x;   // 0 .. 49151
    int lane = id & 31;
    int p    = id >> 14;
    const float* W = (p == 0) ? Wk : (p == 1) ? Wv : Wq;
    int n  = ((id >> 9) & 31) * 16 + (lane & 15);
    int k0 = ((id >> 5) & 15) * 32 + 16 * (lane >> 4);
    const float* src = W + n * DD + k0;
    union { _Float16 h[16]; v4u u[2]; } t;
#pragma unroll
    for (int j = 0; j < 16; ++j) t.h[j] = (_Float16)src[j];
    ((v4u*)Bpre)[id * 2]     = t.u[0];
    ((v4u*)Bpre)[id * 2 + 1] = t.u[1];
}

// Stage one 16KB B chunk (proj p, n-group g, k-chunk kc) into blds buffer buf.
__device__ __forceinline__ void stage_b(const _Float16* __restrict__ Bpre,
                                        _Float16* blds, int tid,
                                        int p, int g, int kc, int buf)
{
    const int ntl  = tid >> 5;
    const int lane = tid & 31;
    _Float16* base = blds + (buf << 13);         // buf * 8192 halfs
#pragma unroll
    for (int s = 0; s < 2; ++s) {
        int kt = kc * 2 + s;
        int u4 = ((((p * 32 + (g * 8 + ntl)) * 16 + kt) * 32) + lane) * 2; // 16B units
        const v4u* src = ((const v4u*)Bpre) + u4;
        _Float16* d = &base[(ntl * 2 + s) * 512 + lane * 16];
#if HAVE_ASYNC_LDS
        __builtin_amdgcn_global_load_async_to_lds_b128(TO_GAS(src), TO_LAS(d), 0, 0);
        __builtin_amdgcn_global_load_async_to_lds_b128(TO_GAS(src), TO_LAS(d), 16, 0);
#else
        v4u u0 = src[0];
        v4u u1 = src[1];
        *(v4u*)d       = u0;
        *(v4u*)(d + 8) = u1;
#endif
    }
}

// ---------------------------------------------------------------------------
// Fused kernel: block = (a, group of 4 batches). Computes k,v,q projections
// with v_wmma_f32_16x16x32_f16, folds them immediately into kf/vf/scores,
// then softmax(rank=8) and out = attn @ vf^T, written straight to d_out.
// ---------------------------------------------------------------------------
__global__ __launch_bounds__(256, 1) void qkv_lowrank_attn(
    const float* __restrict__ x,
    const float* __restrict__ bk, const float* __restrict__ bv,
    const float* __restrict__ bq, const float* __restrict__ factor,
    const _Float16* __restrict__ Bpre, float* __restrict__ out)
{
    extern __shared__ char smem[];
    _Float16* xlds = (_Float16*)(smem + OFF_X);
    _Float16* blds = (_Float16*)(smem + OFF_B);
    float* flds = (float*)(smem + OFF_F);
    float* kfb  = (float*)(smem + OFF_KF);
    float* vfb  = (float*)(smem + OFF_VF);
    float* sc   = (float*)(smem + OFF_SC);
    float* bl   = (float*)(smem + OFF_BIAS);

    const int tid   = threadIdx.x;
    const int lane  = tid & 31;
    const int w     = tid >> 5;          // wave id 0..7  (== m-tile)
    const int a     = blockIdx.x & 63;   // head-slice
    const int bg    = blockIdx.x >> 6;   // batch group (4 batches)
    const int lhalf = lane >> 4;

    // ---- warm L2/WGP$ with first projection's weight slab (global_prefetch) ----
#pragma unroll
    for (int i = 0; i < 8; ++i)
        __builtin_prefetch((const char*)Bpre + (size_t)(tid + (i << 8)) * 256, 0, 1);

    // ---- stage factor[a], biases; zero kf/vf/scores ----
    {
        const v4f* fsrc = (const v4f*)(factor + (size_t)a * (256 * 8));
        ((v4f*)flds)[tid * 2]     = fsrc[tid * 2];
        ((v4f*)flds)[tid * 2 + 1] = fsrc[tid * 2 + 1];
#pragma unroll
        for (int i = 0; i < 2; ++i) {
            int c = tid + 256 * i;
            bl[c] = bk[c]; bl[512 + c] = bv[c]; bl[1024 + c] = bq[c];
        }
#pragma unroll
        for (int i = 0; i < 8; ++i) { kfb[i * 256 + tid] = 0.f; vfb[i * 256 + tid] = 0.f; }
#pragma unroll
        for (int i = 0; i < 32; ++i) sc[i * 256 + tid] = 0.f;
    }
    // ---- stage x tile (128 rows x 512) as f16 into LDS ----
#pragma unroll 1
    for (int it = 0; it < 64; ++it) {
        int idx = tid + (it << 8);           // float4 index 0..16383
        int r   = idx >> 7;                  // row 0..127
        int c   = (idx & 127) << 2;          // col
        int b   = bg * NBB + (r >> 5);
        int l   = a * 32 + (r & 31);
        v4f f4  = *(const v4f*)(x + ((size_t)b * LEN + l) * DD + c);
        union { _Float16 h[4]; v2u u; } t;
        t.h[0] = (_Float16)f4.x; t.h[1] = (_Float16)f4.y;
        t.h[2] = (_Float16)f4.z; t.h[3] = (_Float16)f4.w;
        *(v2u*)&xlds[r * XPITCH + c] = t.u;
    }

    const int arow = w * 16 + (lane & 15);   // A-matrix row for this lane

#pragma unroll 1
    for (int p = 0; p < 3; ++p) {            // 0:k 1:v 2:q
#pragma unroll 1
        for (int g = 0; g < 4; ++g) {        // n-group: cols [g*128, g*128+128)
            v8f acc[8];
            {
                v8f z = {};
#pragma unroll
                for (int nt = 0; nt < 8; ++nt) acc[nt] = z;
            }
            // prologue: stage k-chunk 0 into buffer 0
            stage_b(Bpre, blds, tid, p, g, 0, 0);
            wait_async_copies();
            __syncthreads();
#pragma unroll 1
            for (int kc = 0; kc < 8; ++kc) { // K chunks of 64 (2 wmma k-steps)
                const int buf = kc & 1;
                if (kc < 7)                  // overlap next copy with WMMAs
                    stage_b(Bpre, blds, tid, p, g, kc + 1, buf ^ 1);
                _Float16* bbase = blds + (buf << 13);
#pragma unroll
                for (int s = 0; s < 2; ++s) {
                    int kt = kc * 2 + s;
                    union { v16h v; v4u u[2]; } af;
                    const _Float16* ap = &xlds[arow * XPITCH + kt * 32 + 16 * lhalf];
                    af.u[0] = *(const v4u*)ap;
                    af.u[1] = *(const v4u*)(ap + 8);
#pragma unroll
                    for (int nt = 0; nt < 8; ++nt) {
                        union { v16h v; v4u u[2]; } bf;
                        const _Float16* bp = &bbase[(nt * 2 + s) * 512 + lane * 16];
                        bf.u[0] = *(const v4u*)bp;
                        bf.u[1] = *(const v4u*)(bp + 8);
                        acc[nt] = __builtin_amdgcn_wmma_f32_16x16x32_f16(
                            false, af.v, false, bf.v, (short)0, acc[nt], false, false);
                    }
                }
                wait_async_copies();         // our async writes for kc+1 landed
                __syncthreads();             // all waves done reading buf / writing buf^1
            }
            // ---- fold this (p,g) accumulator slab into kf/vf or scores ----
            const int bb  = w >> 1;
            const int hi  = g >> 1;
            const int lrb = (g & 1) * 128;
            if (p < 2) {
                float kfp[8][8] = {};
#pragma unroll
                for (int nt = 0; nt < 8; ++nt) {
                    int cl  = lane & 15;
                    int col = g * 128 + nt * 16 + cl;
                    int lr  = lrb + nt * 16 + cl;
                    float bsv = bl[p * 512 + col];
                    v4f fa = *(const v4f*)&flds[lr * 8];
                    v4f fb = *(const v4f*)&flds[lr * 8 + 4];
#pragma unroll
                    for (int i = 0; i < 8; ++i) {
                        float val = acc[nt][i] + bsv;
                        kfp[i][0] += val * fa.x; kfp[i][1] += val * fa.y;
                        kfp[i][2] += val * fa.z; kfp[i][3] += val * fa.w;
                        kfp[i][4] += val * fb.x; kfp[i][5] += val * fb.y;
                        kfp[i][6] += val * fb.z; kfp[i][7] += val * fb.w;
                    }
                }
#pragma unroll
                for (int m = 1; m < 16; m <<= 1)
#pragma unroll
                    for (int i = 0; i < 8; ++i)
#pragma unroll
                        for (int r = 0; r < 8; ++r)
                            kfp[i][r] += __shfl_xor(kfp[i][r], m);
                if ((lane & 15) == 0) {
                    float* dst = (p == 0) ? kfb : vfb;
#pragma unroll
                    for (int i = 0; i < 8; ++i) {
                        int h = 2 * ((w & 1) * 16 + i + 8 * lhalf) + hi;
#pragma unroll
                        for (int r = 0; r < 8; ++r)
                            atomicAdd(&dst[(bb * 64 + h) * 8 + r], kfp[i][r]);
                    }
                }
            } else {
                float sp[8][8] = {};            // [nt][r]
#pragma unroll
                for (int i = 0; i < 8; ++i) {
                    int h = 2 * ((w & 1) * 16 + i + 8 * lhalf) + hi;
                    v4f ka  = *(const v4f*)&kfb[(bb * 64 + h) * 8];
                    v4f kb2 = *(const v4f*)&kfb[(bb * 64 + h) * 8 + 4];
#pragma unroll
                    for (int nt = 0; nt < 8; ++nt) {
                        int col = g * 128 + nt * 16 + (lane & 15);
                        float val = acc[nt][i] + bl[1024 + col];
                        sp[nt][0] += val * ka.x;  sp[nt][1] += val * ka.y;
                        sp[nt][2] += val * ka.z;  sp[nt][3] += val * ka.w;
                        sp[nt][4] += val * kb2.x; sp[nt][5] += val * kb2.y;
                        sp[nt][6] += val * kb2.z; sp[nt][7] += val * kb2.w;
                    }
                }
#pragma unroll
                for (int nt = 0; nt < 8; ++nt)
#pragma unroll
                    for (int r = 0; r < 8; ++r)
                        sp[nt][r] += __shfl_xor(sp[nt][r], 16);
                if (lane < 16) {
#pragma unroll
                    for (int nt = 0; nt < 8; ++nt) {
                        int lr = lrb + nt * 16 + lane;
#pragma unroll
                        for (int r = 0; r < 8; ++r)
                            atomicAdd(&sc[(bb * 256 + lr) * 8 + r], 0.125f * sp[nt][r]);
                    }
                }
            }
        }
    }
    __syncthreads();
    // ---- softmax over rank (8) ----
#pragma unroll
    for (int j = 0; j < 4; ++j) {
        int rw = tid + 256 * j;              // 0..1023 rows
        float* s8 = &sc[rw * 8];
        float m = s8[0];
#pragma unroll
        for (int r = 1; r < 8; ++r) m = fmaxf(m, s8[r]);
        float e[8], sum = 0.f;
#pragma unroll
        for (int r = 0; r < 8; ++r) { e[r] = __expf(s8[r] - m); sum += e[r]; }
        float inv = 1.0f / sum;
#pragma unroll
        for (int r = 0; r < 8; ++r) s8[r] = e[r] * inv;
    }
    __syncthreads();
    // ---- out[bb][lr][h] = sum_r attn[lr][r] * vf[h][r], coalesced store ----
#pragma unroll 1
    for (int j = 0; j < 4; ++j) {
        int pr = tid + 256 * j;
        int bb = pr >> 8, lr = pr & 255;
        float at[8];
#pragma unroll
        for (int r = 0; r < 8; ++r) at[r] = sc[pr * 8 + r];
        int b = bg * NBB + bb;
        int orow = a * 32 + (lr >> 3);
        float* dst = out + ((size_t)(b * LEN + orow)) * DD + (lr & 7) * 64;
#pragma unroll 1
        for (int h4 = 0; h4 < 64; h4 += 4) {
            v4f o;
#pragma unroll
            for (int q4 = 0; q4 < 4; ++q4) {
                const float* vr = &vfb[(bb * 64 + h4 + q4) * 8];
                float s = 0.f;
#pragma unroll
                for (int r = 0; r < 8; ++r) s += at[r] * vr[r];
                o[q4] = s;
            }
            *(v4f*)(dst + h4) = o;
        }
    }
}

extern "C" void kernel_launch(void* const* d_in, const int* in_sizes, int n_in,
                              void* d_out, int out_size, void* d_ws, size_t ws_size,
                              hipStream_t stream) {
    const float* x  = (const float*)d_in[0];
    const float* Wq = (const float*)d_in[1];
    const float* bq = (const float*)d_in[2];
    const float* Wk = (const float*)d_in[3];
    const float* bk = (const float*)d_in[4];
    const float* Wv = (const float*)d_in[5];
    const float* bv = (const float*)d_in[6];
    const float* fc = (const float*)d_in[7];
    float* out = (float*)d_out;
    _Float16* Bpre = (_Float16*)d_ws;   // 3*512*512 f16 = 1.5 MB

    (void)in_sizes; (void)n_in; (void)out_size; (void)ws_size;
    (void)hipFuncSetAttribute((const void*)qkv_lowrank_attn,
                              hipFuncAttributeMaxDynamicSharedMemorySize, SMEM_TOTAL);
    prep_weights<<<dim3(192), dim3(256), 0, stream>>>(Wk, Wv, Wq, Bpre);
    qkv_lowrank_attn<<<dim3(512), dim3(256), SMEM_TOTAL, stream>>>(
        x, bk, bv, bq, fc, Bpre, out);
}
